// VanillaCRF_26585847562381
// MI455X (gfx1250) — matrix operand order repaired and checked by reading, not compiled, not run on hardware
//
#include <hip/hip_runtime.h>
#include <hip/hip_bf16.h>

typedef __attribute__((ext_vector_type(16))) _Float16 v16h;
typedef __attribute__((ext_vector_type(8)))  _Float16 v8h;
typedef __attribute__((ext_vector_type(8)))  float    v8f;

#define Pn   64
#define Ln   128
#define Hn   768
#define Gn   3072   // 4*H
#define H2   1536   // 2*H

// ---------------------------------------------------------------------------
// WMMA fragment helpers (CDNA5 16x16x32 f16 layouts, wave32)
// ---------------------------------------------------------------------------
__device__ __forceinline__ v8f wmma_f16(v16h a, v16h b, v8f c) {
  return __builtin_amdgcn_wmma_f32_16x16x32_f16(false, a, false, b, (short)0, c,
                                                false, false);
}

__device__ __forceinline__ v16h make_v16(v8h lo, v8h hi) {
  return __builtin_shufflevector(lo, hi, 0,1,2,3,4,5,6,7,8,9,10,11,12,13,14,15);
}

// A-matrix 16x32 f16: lane m=lane&15, half h=lane>>4.
// VGPRs 0-3 hold K = k0 + 8h + {0..7}, VGPRs 4-7 hold K = k0 + 8h + 16 + {0..7}.
__device__ __forceinline__ v16h glb_load_a(const _Float16* base, long rs, int m0, int k0) {
  int lane = threadIdx.x & 31;
  long m = (long)((lane & 15) + m0);
  int hh = lane >> 4;
  const _Float16* p0 = base + m * rs + k0 + hh * 8;
  return make_v16(*(const v8h*)p0, *(const v8h*)(p0 + 16));
}

// Same A fragment but sourced from LDS-resident h (row-major 64x768 f16).
__device__ __forceinline__ v16h lds_load_a(const _Float16* h, int m0, int k0) {
  int lane = threadIdx.x & 31;
  int m = (lane & 15) + m0, hh = lane >> 4;
  const _Float16* p0 = h + m * Hn + k0 + hh * 8;
  return make_v16(*(const v8h*)p0, *(const v8h*)(p0 + 16));
}

// B-matrix 32x16 f16 from row-major weight W[n][k] (B = W^T):
// lane n=lane&15, half h=lane>>4; 8 f16 of row n at k0+8h, 8 at k0+16+8h.
__device__ __forceinline__ v16h glb_load_b(const _Float16* w, long rs, int n0, int k0) {
  int lane = threadIdx.x & 31;
  long n = (long)((lane & 15) + n0);
  int hh = lane >> 4;
  const _Float16* p = w + n * rs + k0 + hh * 8;
  return make_v16(*(const v8h*)p, *(const v8h*)(p + 16));
}

__device__ __forceinline__ float sigmoidf(float x) { return 1.f / (1.f + __expf(-x)); }
__device__ __forceinline__ float lse2(float a, float b) {
  float m = fmaxf(a, b);
  return m + __logf(__expf(a - m) + __expf(b - m));
}

// ---------------------------------------------------------------------------
// Prep kernels
// ---------------------------------------------------------------------------
__global__ void k_f32_to_f16(const float* __restrict__ src, _Float16* __restrict__ dst, int n) {
  int i = blockIdx.x * blockDim.x + threadIdx.x;
  if (i < n) dst[i] = (_Float16)src[i];
}

__global__ void k_build_seq(const float* __restrict__ user, const float* __restrict__ sys,
                            _Float16* __restrict__ seqf) {
  int idx = blockIdx.x * blockDim.x + threadIdx.x;
  if (idx >= Ln * Hn) return;
  int t = idx / Hn, j = idx - t * Hn;
  const float* src = (t & 1) ? sys : user;
  seqf[idx] = (_Float16)src[(t >> 1) * Hn + j];
}

// ---------------------------------------------------------------------------
// xw0 = seq(128x768) @ Wih0^T(768x3072) + b   (shared across all P)
// ---------------------------------------------------------------------------
__global__ __launch_bounds__(512) void k_gemm_xw0(const _Float16* __restrict__ A,
                                                  const _Float16* __restrict__ W,
                                                  const float* __restrict__ bias,
                                                  float* __restrict__ out) {
  int w = blockIdx.x * 16 + (threadIdx.x >> 5);
  if (w >= (Ln / 16) * (Gn / 16)) return;   // wave-uniform
  int mt = (w & 7) * 16, nt = (w >> 3) * 16;
  v8f acc = {};
  for (int k = 0; k < Hn; k += 32) {
    v16h a = glb_load_a(A, Hn, mt, k);
    v16h b = glb_load_b(W, Hn, nt, k);
    acc = wmma_f16(a, b, acc);
  }
  int lane = threadIdx.x & 31;
  int n = (lane & 15) + nt, hh = lane >> 4;
  float bv = bias[n];
#pragma unroll
  for (int r = 0; r < 8; ++r)
    out[(long)(mt + r + 8 * hh) * Gn + n] = acc[r] + bv;
}

// ---------------------------------------------------------------------------
// Fused LSTM cell applied to 4 in-register gate accumulators.
// Block id -> (m-tile, j-tile) is stable across steps, so the c-state in
// global cbuf is owner-exclusive (no sync needed).
// ---------------------------------------------------------------------------
__device__ __forceinline__ void lstm_cell_tile(const v8f acc[4],
                                               float x0, float x1, float x2, float x3,
                                               int mt, int j, int t, int net,
                                               const _Float16* hcur, _Float16* hnxt,
                                               float* __restrict__ cst,
                                               _Float16* __restrict__ out) {
  int lane = threadIdx.x & 31;
  int hh = lane >> 4;
#pragma unroll
  for (int r = 0; r < 8; ++r) {
    int p = mt + r + 8 * hh;
    long cidx = (long)p * Hn + j;
    float c2 = sigmoidf(acc[1][r] + x1) * cst[cidx] +
               sigmoidf(acc[0][r] + x0) * tanhf(acc[2][r] + x2);
    float h2 = sigmoidf(acc[3][r] + x3) * tanhf(c2);
    int lim = (net == 0) ? 2 * (p + 1) : 2 * p + 1;
    bool msk = t < lim;
    float hp = (float)hcur[cidx];
    float hn = msk ? h2 : hp;
    if (msk) cst[cidx] = c2;
    hnxt[cidx] = (_Float16)hn;
    out[((long)p * Ln + t) * H2 + j] = (_Float16)hn;
  }
}

// ---------------------------------------------------------------------------
// Layer-0 scan. One block per (net,dir) instance; h ping-pong in LDS.
// Per step, each wave owns 6 (m-tile, j-tile) blocks and computes the four
// gate tiles i/f/g/o for that block in registers, then applies the cell.
// One __syncthreads() per step (GEMM reads h[s&1], cell writes h[(s+1)&1]).
// ---------------------------------------------------------------------------
__global__ __launch_bounds__(1024) void k_scan_l0(const _Float16* __restrict__ whh0f,
                                                  const float* __restrict__ xw0,
                                                  float* __restrict__ cbuf,
                                                  _Float16* __restrict__ x01_all) {
  extern __shared__ _Float16 hlds[];           // 2 * 64*768 f16 = 192 KB
  int inst = blockIdx.x;                       // 0..3 : net*2+dir
  int net = inst >> 1, dir = inst & 1;
  const _Float16* W = whh0f + (long)inst * Gn * Hn;
  const float* xw = xw0 + (long)inst * Ln * Gn;
  float* cst = cbuf + (long)inst * Pn * Hn;
  _Float16* out = x01_all + (long)net * Pn * Ln * H2 + dir * Hn;
  int tid = threadIdx.x, w = tid >> 5;
  int lane = tid & 31, n = lane & 15;

  for (int i = tid; i < Pn * Hn; i += 1024) { hlds[i] = (_Float16)0.f; cst[i] = 0.f; }
  __syncthreads();

  for (int s = 0; s < Ln; ++s) {
    int t = dir ? (Ln - 1 - s) : s;
    const _Float16* hcur = hlds + (s & 1) * (Pn * Hn);
    _Float16* hnxt = hlds + ((s + 1) & 1) * (Pn * Hn);
#pragma unroll 1
    for (int b = 0; b < 6; ++b) {              // 192 blocks / 32 waves
      int id = w * 6 + b;
      int mt = (id & 3) * 16, j0 = (id >> 2) * 16;
      v8f acc[4] = {{}, {}, {}, {}};
      for (int k = 0; k < Hn; k += 32) {
        v16h a = lds_load_a(hcur, mt, k);
#pragma unroll
        for (int q = 0; q < 4; ++q) {
          v16h bb = glb_load_b(W, Hn, q * Hn + j0, k);
          acc[q] = wmma_f16(a, bb, acc[q]);
        }
      }
      int j = j0 + n;
      const float* xr = xw + (long)t * Gn + j;
      lstm_cell_tile(acc, xr[0], xr[Hn], xr[2 * Hn], xr[3 * Hn],
                     mt, j, t, net, hcur, hnxt, cst, out);
    }
    __syncthreads();
  }
}

// ---------------------------------------------------------------------------
// Layer-1 scan, fused input projection: K = 1536 (x_t from x01, global)
// + 768 (h, LDS).  gates = [x_t , h] @ [Wih1 ; Whh1]^T + b1.
// ---------------------------------------------------------------------------
__global__ __launch_bounds__(1024) void k_scan_l1(const _Float16* __restrict__ wih1f,
                                                  const _Float16* __restrict__ whh1f,
                                                  const float* __restrict__ post_b1,
                                                  const float* __restrict__ prior_b1,
                                                  const _Float16* __restrict__ x01_all,
                                                  float* __restrict__ cbuf,
                                                  _Float16* __restrict__ x11_all) {
  extern __shared__ _Float16 hlds[];
  int inst = blockIdx.x;
  int net = inst >> 1, dir = inst & 1;
  const _Float16* W0 = wih1f + (long)inst * Gn * H2;   // Wih1: 3072x1536
  const _Float16* W1 = whh1f + (long)inst * Gn * Hn;   // Whh1: 3072x768
  const float* bias = ((net == 0) ? post_b1 : prior_b1) + dir * Gn;
  const _Float16* xin = x01_all + (long)net * Pn * Ln * H2;
  float* cst = cbuf + (long)inst * Pn * Hn;
  _Float16* out = x11_all + (long)net * Pn * Ln * H2 + dir * Hn;
  int tid = threadIdx.x, w = tid >> 5;
  int lane = tid & 31, n = lane & 15;

  for (int i = tid; i < Pn * Hn; i += 1024) { hlds[i] = (_Float16)0.f; cst[i] = 0.f; }
  __syncthreads();

  for (int s = 0; s < Ln; ++s) {
    int t = dir ? (Ln - 1 - s) : s;
    const _Float16* hcur = hlds + (s & 1) * (Pn * Hn);
    _Float16* hnxt = hlds + ((s + 1) & 1) * (Pn * Hn);
    const _Float16* a0 = xin + (long)t * H2;           // row m adds m*(Ln*H2)
#pragma unroll 1
    for (int b = 0; b < 6; ++b) {
      int id = w * 6 + b;
      int mt = (id & 3) * 16, j0 = (id >> 2) * 16;
      v8f acc[4] = {{}, {}, {}, {}};
      for (int k = 0; k < H2; k += 32) {               // input-projection part
        v16h a = glb_load_a(a0, (long)Ln * H2, mt, k);
#pragma unroll
        for (int q = 0; q < 4; ++q) {
          v16h bb = glb_load_b(W0, H2, q * Hn + j0, k);
          acc[q] = wmma_f16(a, bb, acc[q]);
        }
      }
      for (int k = 0; k < Hn; k += 32) {               // recurrent part
        v16h a = lds_load_a(hcur, mt, k);
#pragma unroll
        for (int q = 0; q < 4; ++q) {
          v16h bb = glb_load_b(W1, Hn, q * Hn + j0, k);
          acc[q] = wmma_f16(a, bb, acc[q]);
        }
      }
      int j = j0 + n;
      lstm_cell_tile(acc, bias[j], bias[Hn + j], bias[2 * Hn + j], bias[3 * Hn + j],
                     mt, j, t, net, hcur, hnxt, cst, out);
    }
    __syncthreads();
  }
}

// ---------------------------------------------------------------------------
// em[p][t][c] = x11_post[p][t][:] . post_proj_W[c] + b[c]
// ---------------------------------------------------------------------------
__global__ void k_proj_em(const _Float16* __restrict__ x11_post,
                          const float* __restrict__ Wp, const float* __restrict__ bp,
                          float* __restrict__ em) {
  int idx = blockIdx.x * blockDim.x + threadIdx.x;
  if (idx >= Pn * Ln * 2) return;
  int c = idx & 1;
  long row = idx >> 1;
  const _Float16* xr = x11_post + row * H2;
  const float* wr = Wp + c * H2;
  float s = bp[c];
  for (int k = 0; k < H2; ++k) s += (float)xr[k] * wr[k];
  em[idx] = s;
}

// ---------------------------------------------------------------------------
// CRF forward + gold score, C=2; one thread per p, reduce to loss_crf.
// ---------------------------------------------------------------------------
__global__ __launch_bounds__(64) void k_crf(const float* __restrict__ em,
                                            const float* __restrict__ trans,
                                            const int* __restrict__ userlab,
                                            const int* __restrict__ syslab,
                                            float* __restrict__ out) {
  __shared__ float red[64];
  __shared__ int labs[Ln];
  int tid = threadIdx.x;
  labs[2 * tid] = userlab[tid];
  labs[2 * tid + 1] = syslab[tid];
  __syncthreads();
  int p = tid, l = 2 * (p + 1);
  float t00 = trans[0], t01 = trans[1], t10 = trans[2], t11 = trans[3];
  const float* emp = em + (long)p * Ln * 2;
  float a0 = emp[0], a1 = emp[1];
  float gold = emp[labs[0]];
  for (int t = 1; t < Ln; ++t) {
    if (t < l) {
      float n0 = emp[t * 2 + 0] + lse2(a0 + t00, a1 + t10);
      float n1 = emp[t * 2 + 1] + lse2(a0 + t01, a1 + t11);
      a0 = n0; a1 = n1;
      gold += emp[t * 2 + labs[t]] + trans[labs[t - 1] * 2 + labs[t]];
    }
  }
  red[p] = lse2(a0, a1) - gold;
  __syncthreads();
  for (int off = 32; off > 0; off >>= 1) {
    if (tid < off) red[tid] += red[tid + off];
    __syncthreads();
  }
  if (tid == 0) out[0] = red[0] / (float)Pn;
}

// ---------------------------------------------------------------------------
// prior_emis from x11_prior[p][l-2], MSE vs em[p][l-1]; writes out[1], out[2].
// ---------------------------------------------------------------------------
__global__ __launch_bounds__(128) void k_mle(const _Float16* __restrict__ x11_prior,
                                             const float* __restrict__ Wpr,
                                             const float* __restrict__ bpr,
                                             const float* __restrict__ em,
                                             float* __restrict__ out) {
  __shared__ float red[128];
  int tid = threadIdx.x, p = tid >> 1, c = tid & 1;
  const _Float16* xr = x11_prior + ((long)p * Ln + 2 * p) * H2;   // l-2 = 2p
  const float* wr = Wpr + c * H2;
  float s = bpr[c];
  for (int k = 0; k < H2; ++k) s += (float)xr[k] * wr[k];
  float pl = em[((long)p * Ln + (2 * p + 1)) * 2 + c];            // l-1 = 2p+1
  float d = s - pl;
  red[tid] = d * d;
  __syncthreads();
  for (int off = 64; off > 0; off >>= 1) {
    if (tid < off) red[tid] += red[tid + off];
    __syncthreads();
  }
  if (tid == 0) {
    float lm = red[0] / 128.f;
    out[1] = lm;
    out[2] = out[0] + 0.1f * lm;
  }
}

// ---------------------------------------------------------------------------
extern "C" void kernel_launch(void* const* d_in, const int* in_sizes, int n_in,
                              void* d_out, int out_size, void* d_ws, size_t ws_size,
                              hipStream_t stream) {
  const float* user   = (const float*)d_in[0];
  const float* sysr   = (const float*)d_in[1];
  const float* pWih0  = (const float*)d_in[2];
  const float* pWhh0  = (const float*)d_in[3];
  const float* pB0    = (const float*)d_in[4];
  const float* pWih1  = (const float*)d_in[5];
  const float* pWhh1  = (const float*)d_in[6];
  const float* pB1    = (const float*)d_in[7];
  const float* rWih0  = (const float*)d_in[8];
  const float* rWhh0  = (const float*)d_in[9];
  const float* rB0    = (const float*)d_in[10];
  const float* rWih1  = (const float*)d_in[11];
  const float* rWhh1  = (const float*)d_in[12];
  const float* rB1    = (const float*)d_in[13];
  const float* projW  = (const float*)d_in[14];
  const float* projB  = (const float*)d_in[15];
  const float* prjWr  = (const float*)d_in[16];
  const float* prjBr  = (const float*)d_in[17];
  const float* trans  = (const float*)d_in[18];
  const int*   ulab   = (const int*)d_in[19];
  const int*   slab   = (const int*)d_in[20];
  float* out = (float*)d_out;

  char* base = (char*)d_ws;
  size_t off = 0;
  auto alloc = [&](size_t bytes) { size_t r = off; off = (off + bytes + 255) & ~(size_t)255; return r; };

  const long nWih0 = 2L * Gn * Hn, nWhh0 = 2L * Gn * Hn;
  const long nWih1 = 2L * Gn * H2, nWhh1 = 2L * Gn * Hn;

  _Float16* seqf  = (_Float16*)(base + alloc((size_t)Ln * Hn * 2));
  _Float16* wih0f = (_Float16*)(base + alloc((size_t)2 * nWih0 * 2));
  _Float16* whh0f = (_Float16*)(base + alloc((size_t)2 * nWhh0 * 2));
  _Float16* wih1f = (_Float16*)(base + alloc((size_t)2 * nWih1 * 2));
  _Float16* whh1f = (_Float16*)(base + alloc((size_t)2 * nWhh1 * 2));
  float*    xw0   = (float*)   (base + alloc((size_t)4 * Ln * Gn * 4));
  _Float16* x01   = (_Float16*)(base + alloc((size_t)2 * Pn * Ln * H2 * 2));
  _Float16* x11   = (_Float16*)(base + alloc((size_t)2 * Pn * Ln * H2 * 2));
  float*    cbuf  = (float*)   (base + alloc((size_t)4 * Pn * Hn * 4));
  float*    em    = (float*)   (base + alloc((size_t)Pn * Ln * 2 * 4));

  auto cvt = [&](const float* s, _Float16* d, long n) {
    k_f32_to_f16<<<(int)((n + 255) / 256), 256, 0, stream>>>(s, d, (int)n);
  };
  k_build_seq<<<(Ln * Hn + 255) / 256, 256, 0, stream>>>(user, sysr, seqf);
  cvt(pWih0, wih0f,          nWih0);  cvt(rWih0, wih0f + nWih0, nWih0);
  cvt(pWhh0, whh0f,          nWhh0);  cvt(rWhh0, whh0f + nWhh0, nWhh0);
  cvt(pWih1, wih1f,          nWih1);  cvt(rWih1, wih1f + nWih1, nWih1);
  cvt(pWhh1, whh1f,          nWhh1);  cvt(rWhh1, whh1f + nWhh1, nWhh1);

  // xw0 per instance (inst = net*2+dir), bias folded in
  for (int inst = 0; inst < 4; ++inst) {
    int net = inst >> 1, dir = inst & 1;
    const float* bias = (net ? rB0 : pB0) + dir * Gn;
    k_gemm_xw0<<<96, 512, 0, stream>>>(seqf, wih0f + (long)inst * Gn * Hn, bias,
                                       xw0 + (long)inst * Ln * Gn);
  }

  const size_t hbytes = (size_t)2 * Pn * Hn * sizeof(_Float16);   // 192 KB LDS
  k_scan_l0<<<4, 1024, hbytes, stream>>>(whh0f, xw0, cbuf, x01);
  k_scan_l1<<<4, 1024, hbytes, stream>>>(wih1f, whh1f, pB1, rB1, x01, cbuf, x11);

  k_proj_em<<<(Pn * Ln * 2 + 255) / 256, 256, 0, stream>>>(x11, projW, projB, em);
  k_crf<<<1, 64, 0, stream>>>(em, trans, ulab, slab, out);
  k_mle<<<1, 128, 0, stream>>>(x11 + (long)Pn * Ln * H2, prjWr, prjBr, em, out);

  (void)in_sizes; (void)n_in; (void)out_size; (void)ws_size;
}
